// DimeNetPlusPlus_64080912056988
// MI455X (gfx1250) — matrix operand order repaired and checked by reading, not compiled, not run on hardware
//
#include <hip/hip_runtime.h>
#include <hip/hip_bf16.h>
#include <stdint.h>

// ---------------- problem dimensions (compile-time) ----------------
#define BSZ   2
#define NNODE 4096
#define NEDGE 65536
#define EDGE_SHIFT 16          // log2(NEDGE)
#define NTRIP 524288
#define HC    128
#define OCw   1
#define NBLK  4
#define IEw   64
#define BEw   8
#define NSw   7
#define NRw   6
#define OEw   256

// ---------------- WMMA types (gfx1250, wave32) ----------------
typedef __bf16 bf16_t;
typedef bf16_t v16bf __attribute__((ext_vector_type(16)));
typedef float  v8f   __attribute__((ext_vector_type(8)));

__device__ __forceinline__ unsigned short f32_to_bf16(float f) {
  union { float f; unsigned int u; } v; v.f = f;
  unsigned int r = v.u + 0x7FFFu + ((v.u >> 16) & 1u);   // RNE
  return (unsigned short)(r >> 16);
}
// fast SiLU: v_exp_f32 + v_rcp_f32 + v_mul (no IEEE-div fixup chain)
__device__ __forceinline__ float silu_f(float v) {
  return v * __builtin_amdgcn_rcpf(1.0f + __expf(-v));
}

// epilogue feature bits
enum { E_BIAS = 1, E_PRE = 2, E_CMUL = 4, E_ACT = 8, E_RES = 16 };

// ---------------- weight pack: W[K,N] f32 -> bf16 B-fragments ----------------
// B fragment (contiguous k order used here, same as A in sA): lane L holds col n = L&15, element e holds K = (L>>4)*16 + e.
// pW[((nt*nK + kc)*32 + lane)*16 + e]
__global__ void pack_w_kernel(const float* __restrict__ W,
                              unsigned short* __restrict__ pW,
                              int K, int N, int nK, int nN) {
  int gid = blockIdx.x * blockDim.x + threadIdx.x;
  int total = nN * nK * 512;
  if (gid >= total) return;
  int e    = gid & 15;
  int lane = (gid >> 4) & 31;
  int kc   = (gid >> 9) % nK;
  int nt   = gid / (nK * 512);
  int k = kc * 32 + ((lane >> 4) << 4) + e;
  int n = nt * 16 + (lane & 15);
  float v = (k < K && n < N) ? W[(long)k * N + n] : 0.0f;
  pW[gid] = f32_to_bf16(v);
}

// ---------------- bf16-WMMA GEMM, fully specialized ----------------
// C[M,N] = epilogue( A[M,K] @ W[K,N] );  W pre-packed.  M must be multiple of 64.
// GATHER: arow = (m>>16)*NNODE + aIdx[m & 0xFFFF]     (xe gather only)
// AMUL:   A multiplied elementwise by amul[m,K]
// EPI:    v = acc (+bias[n]) (+pre[m,n]) (*cmul[m,n]) (silu) (+resid[m,n])
#define BM 64
#define BN 64

template<int K, int N, bool GATHER, bool AMUL, int EPI>
__global__ __launch_bounds__(256)
void wmma_gemm(const float* __restrict__ A, const unsigned short* __restrict__ pW,
               const float* __restrict__ bias, const float* __restrict__ pre,
               const float* __restrict__ resid, const float* __restrict__ amul,
               const float* __restrict__ cmul, float* __restrict__ C,
               const int* __restrict__ aIdx)
{
  constexpr int nK = (K + 31) / 32;
  constexpr int GPR = nK * 8;                 // 4-wide k-groups per row
  constexpr bool FULLN = (N % BN) == 0;       // all launched columns in range
  // whole A block-tile, fragment-major: sA[(((kc*4 + wm)*32 + lane)*16 + e]
  __shared__ __align__(32) unsigned short sA[nK * 4 * 32 * 16];

  const int tid  = threadIdx.x;
  const int lane = tid & 31;
  const int wave = tid >> 5;        // 8 waves
  const int wm   = wave & 3;        // 4 tiles in M
  const int wg   = wave >> 2;       // 2 groups of 2 N-tiles
  const int half = lane >> 4;
  const int l16  = lane & 15;
  const int m0   = blockIdx.y * BM;
  const int n0   = blockIdx.x * BN;
  const int nt0  = blockIdx.x * 4 + wg * 2;

  // ---- stage full A tile (64 x nK*32), fragment-major, 4 k per thread ----
#pragma unroll
  for (int it = 0; it < 2 * nK; ++it) {
    int idx = tid + it * 256;
    int r = idx / GPR;                 // tile row 0..63 (GPR is a power of two)
    int k = (idx & (GPR - 1)) << 2;    // 0..nK*32-4
    int gm = m0 + r;
    float4 v = make_float4(0.f, 0.f, 0.f, 0.f);
    long arow;
    if (GATHER) arow = (long)(gm >> EDGE_SHIFT) * NNODE + aIdx[gm & (NEDGE - 1)];
    else        arow = gm;
    if (K % 4 == 0) {
      if (k < K) {
        v = *(const float4*)(A + arow * (long)K + k);
        if (AMUL) {
          const float4 mm = *(const float4*)(amul + (long)gm * K + k);
          v.x *= mm.x; v.y *= mm.y; v.z *= mm.z; v.w *= mm.w;
        }
        if (!GATHER && !AMUL) {
          // prefetch next M-block's tile (speculative; dropped if OOB)
          __builtin_prefetch(A + (arow + BM) * (long)K + k, 0, 1);
        }
      }
    } else {
      long base = arow * (long)K + k;
      if (k     < K) v.x = A[base];
      if (k + 1 < K) v.y = A[base + 1];
      if (k + 2 < K) v.z = A[base + 2];
      if (k + 3 < K) v.w = A[base + 3];
    }
    int kc    = k >> 5;
    int kk    = k & 31;
    int dhalf = kk >> 4;
    int e0    = kk & 15;
    int dlane = dhalf * 16 + (r & 15);
    int dwm   = r >> 4;
    unsigned int lo = (unsigned int)f32_to_bf16(v.x) | ((unsigned int)f32_to_bf16(v.y) << 16);
    unsigned int hi = (unsigned int)f32_to_bf16(v.z) | ((unsigned int)f32_to_bf16(v.w) << 16);
    *(uint2*)&sA[(((kc * 4) + dwm) * 32 + dlane) * 16 + e0] = make_uint2(lo, hi);
  }
  __syncthreads();

  // ---- k-loop: pure ds_load_b128 + global_load_b128 + WMMA stream ----
  v8f acc0 = {}, acc1 = {};
#pragma unroll
  for (int kc = 0; kc < nK; ++kc) {
    v16bf af  = *(const v16bf*)&sA[(((kc * 4) + wm) * 32 + lane) * 16];
    v16bf bf0 = *(const v16bf*)(pW + (((long)nt0       * nK + kc) * 32 + lane) * 16);
    v16bf bf1 = *(const v16bf*)(pW + (((long)(nt0 + 1) * nK + kc) * 32 + lane) * 16);
    acc0 = __builtin_amdgcn_wmma_f32_16x16x32_bf16(false, af, false, bf0, (short)0, acc0, false, false);
    acc1 = __builtin_amdgcn_wmma_f32_16x16x32_bf16(false, af, false, bf1, (short)0, acc1, false, false);
  }

  // ---- epilogue: C/D row = half*8 + vv, col = l16 ----
  const int nA = n0 + wg * 32 + l16;
  const int nB = nA + 16;
  float bA = 0.0f, bB = 0.0f;
  if (EPI & E_BIAS) {
    if (FULLN || nA < N) bA = bias[nA];
    if (FULLN || nB < N) bB = bias[nB];
  }
#pragma unroll
  for (int vv = 0; vv < 8; ++vv) {
    int m = m0 + wm * 16 + half * 8 + vv;
    if (FULLN || nA < N) {
      long off = (long)m * N + nA;
      float val = acc0[vv];
      if (EPI & E_BIAS) val += bA;
      if (EPI & E_PRE)  val += pre[off];
      if (EPI & E_CMUL) val *= cmul[off];
      if (EPI & E_ACT)  val = silu_f(val);
      if (EPI & E_RES)  val += resid[off];
      C[off] = val;
    }
    if (FULLN || nB < N) {
      long off = (long)m * N + nB;
      float val = acc1[vv];
      if (EPI & E_BIAS) val += bB;
      if (EPI & E_PRE)  val += pre[off];
      if (EPI & E_CMUL) val *= cmul[off];
      if (EPI & E_ACT)  val = silu_f(val);
      if (EPI & E_RES)  val += resid[off];
      C[off] = val;
    }
  }
}

// ---------------- small helper kernels ----------------
__global__ void zero_kernel(float* p, long n) {
  long i = (long)blockIdx.x * blockDim.x + threadIdx.x;
  if (i < n) p[i] = 0.0f;
}

__global__ void bessel_kernel(const float* __restrict__ dist,
                              const float* __restrict__ freq,
                              float* __restrict__ rbf, int ME) {
  int m = blockIdx.x * blockDim.x + threadIdx.x;
  if (m >= ME) return;
  float d = dist[m] * 0.2f;
  float x5 = d * d; x5 *= x5; x5 *= d;                 // d^5
  float env = __builtin_amdgcn_rcpf(d) - 28.0f * x5 + 48.0f * x5 * d - 21.0f * x5 * d * d;
  if (d >= 1.0f) env = 0.0f;
#pragma unroll
  for (int r = 0; r < NRw; ++r)
    rbf[(long)m * NRw + r] = env * __sinf(freq[r] * d);
}

__global__ void embed_x_kernel(const float* __restrict__ x,
                               const float* __restrict__ W,
                               const float* __restrict__ b,
                               float* __restrict__ xe, long total) {
  long g = (long)blockIdx.x * blockDim.x + threadIdx.x;
  if (g >= total) return;
  int m = (int)(g / HC), c = (int)(g % HC);
  xe[g] = x[m] * W[c] + b[c];
}

__global__ void scatter_node_kernel(const float* __restrict__ t,
                                    const int* __restrict__ iidx,
                                    float* node) {
  long g = (long)blockIdx.x * blockDim.x + threadIdx.x;
  const int c4 = HC / 4;
  long total = (long)BSZ * NEDGE * c4;
  if (g >= total) return;
  int c = (int)(g % c4) * 4;
  long be = g / c4;
  int e = (int)(be & (NEDGE - 1));
  int b = (int)(be >> EDGE_SHIFT);
  const float4 v = *(const float4*)(t + ((long)(b * NEDGE + e)) * HC + c);
  float* dst = node + ((long)b * NNODE + iidx[e]) * HC + c;
  atomicAdd(dst + 0, v.x); atomicAdd(dst + 1, v.y);
  atomicAdd(dst + 2, v.z); atomicAdd(dst + 3, v.w);
}

__global__ void triplet_scatter_kernel(const float* __restrict__ xkj,
                                       const float* __restrict__ sbfe,
                                       const int* __restrict__ idx_kj,
                                       const int* __restrict__ idx_ji,
                                       float* agg) {
  long g = (long)blockIdx.x * blockDim.x + threadIdx.x;
  const int c4 = IEw / 4;
  long total = (long)BSZ * NTRIP * c4;
  if (g >= total) return;
  int c = (int)(g % c4) * 4;
  long bt = g / c4;
  int t = (int)(bt % NTRIP);
  int b = (int)(bt / NTRIP);
  int ek = idx_kj[t], ej = idx_ji[t];
  const float4 s  = *(const float4*)(sbfe + ((long)(b * NTRIP + t)) * IEw + c);
  const float4 xk = *(const float4*)(xkj  + ((long)(b * NEDGE + ek)) * IEw + c);
  float* dst = agg + ((long)(b * NEDGE + ej)) * IEw + c;
  atomicAdd(dst + 0, s.x * xk.x); atomicAdd(dst + 1, s.y * xk.y);
  atomicAdd(dst + 2, s.z * xk.z); atomicAdd(dst + 3, s.w * xk.w);
}

__global__ void out_dot_kernel(const float* __restrict__ u,
                               const float* __restrict__ W,
                               float* P, int MN, int K) {
  int m = blockIdx.x * blockDim.x + threadIdx.x;
  if (m >= MN) return;
  float s = 0.0f;
  for (int k = 0; k < K; ++k) s += u[(long)m * K + k] * W[k];
  P[m] += s;
}

// ---------------- host-side templated GEMM runner ----------------
template<int K, int N, bool GATHER, bool AMUL, int EPI>
static void run_gemm(const float* A, const float* Wm, const float* bias,
                     const float* pre, const float* resid, const float* amul,
                     const float* cmul, float* C, long M, const int* aIdx,
                     unsigned short* pWbuf, hipStream_t stream) {
  constexpr int nK = (K + 31) / 32;
  constexpr int nN = 4 * ((N + 63) / 64);
  constexpr int ptotal = nN * nK * 512;
  pack_w_kernel<<<dim3((ptotal + 255) / 256), dim3(256), 0, stream>>>(Wm, pWbuf, K, N, nK, nN);
  dim3 grid((unsigned)((N + BN - 1) / BN), (unsigned)(M / BM));
  wmma_gemm<K, N, GATHER, AMUL, EPI><<<grid, dim3(256), 0, stream>>>(
      A, pWbuf, bias, pre, resid, amul, cmul, C, aIdx);
}

// ---------------- host orchestration ----------------
struct BlkP {
  const float *al1W,*al1b,*al2W,*al2b,*bl1W,*bl1b,*bl2W,*bl2b,
              *downW,*jiW,*jib,*kjW,*kjb,*linW,*linb,
              *rbf1W,*rbf2W,*sbf1W,*sbf2W,*upW;
};
struct OutP { const float *linW,*l0W,*l0b,*l1W,*l1b,*l2W,*l2b,*rbfW,*upW; };

extern "C" void kernel_launch(void* const* d_in, const int* in_sizes, int n_in,
                              void* d_out, int out_size, void* d_ws, size_t ws_size,
                              hipStream_t stream)
{
  const float* x      = (const float*)d_in[0];
  const float* dist   = (const float*)d_in[1];
  const float* sbf    = (const float*)d_in[2];
  const int*   iIdx   = (const int*)d_in[3];
  const int*   jIdx   = (const int*)d_in[4];
  const int*   idx_kj = (const int*)d_in[5];
  const int*   idx_ji = (const int*)d_in[6];

  // params flattened in jax pytree order: blocks(4x20), emb(6), freq(1), outs(5x9)
  int p = 7;
  BlkP blk[NBLK];
  for (int b = 0; b < NBLK; ++b) {
    const float** f = (const float**)&blk[b];
    for (int q = 0; q < 20; ++q) f[q] = (const float*)d_in[p++];
  }
  const float* embW  = (const float*)d_in[p++];
  const float* embb  = (const float*)d_in[p++];
  const float* elinW = (const float*)d_in[p++];
  const float* elinb = (const float*)d_in[p++];
  const float* erbfW = (const float*)d_in[p++];
  const float* erbfb = (const float*)d_in[p++];
  const float* freq  = (const float*)d_in[p++];
  OutP outp[NBLK + 1];
  for (int o = 0; o < NBLK + 1; ++o) {
    const float** f = (const float**)&outp[o];
    for (int q = 0; q < 9; ++q) f[q] = (const float*)d_in[p++];
  }

  const long ME = (long)BSZ * NEDGE;   // 131072
  const long MT = (long)BSZ * NTRIP;   // 1048576
  const long MN = (long)BSZ * NNODE;   // 8192

  // workspace carve (f32 units; pW first for 32B alignment)
  float* w = (float*)d_ws;
  unsigned short* pWbuf = (unsigned short*)w; w += 65536 / 2;   // 16*8*512 bf16 max
  float* rbf    = w; w += ME * NRw;
  float* hbuf   = w; w += ME * HC;
  float* tA     = w; w += ME * HC;
  float* tB     = w; w += ME * HC;
  float* tCb    = w; w += ME * HC;
  float* eIE1   = w; w += ME * IEw;
  float* eIE2   = w; w += ME * IEw;
  float* eBE    = w; w += ME * BEw;
  float* tBE    = w; w += MT * BEw;
  float* tIE    = w; w += MT * IEw;
  float* xe     = w; w += MN * HC;
  float* nodeHC = w; w += MN * HC;
  float* nOE1   = w; w += MN * OEw;
  float* nOE2   = w; w += MN * OEw;
  (void)ws_size; (void)n_in; (void)in_sizes; (void)out_size;

  auto zero = [&](float* ptr, long n) {
    zero_kernel<<<dim3((unsigned)((n + 255) / 256)), dim3(256), 0, stream>>>(ptr, n);
  };

  // ---- P = 0 ----
  zero((float*)d_out, MN * OCw);

  // ---- Bessel basis + node embedding ----
  bessel_kernel<<<dim3((unsigned)((ME + 255) / 256)), dim3(256), 0, stream>>>(dist, freq, rbf, (int)ME);
  embed_x_kernel<<<dim3((unsigned)((MN * HC + 255) / 256)), dim3(256), 0, stream>>>(x, embW, embb, xe, MN * HC);

  // rbf_h = silu(rbf @ Wrbf + b) -> tA
  run_gemm<NRw, HC, false, false, E_BIAS | E_ACT>(rbf, erbfW, erbfb, 0, 0, 0, 0, tA, ME, nullptr, pWbuf, stream);
  // h = silu([xe[i] | xe[j] | rbf_h] @ Wlin + b) as 3 K-segment GEMMs
  run_gemm<HC, HC, true, false, 0>(xe, elinW, 0, 0, 0, 0, 0, tB, ME, iIdx, pWbuf, stream);
  run_gemm<HC, HC, true, false, E_PRE>(xe, elinW + HC * HC, 0, tB, 0, 0, 0, tB, ME, jIdx, pWbuf, stream);
  run_gemm<HC, HC, false, false, E_BIAS | E_PRE | E_ACT>(tA, elinW + 2 * HC * HC, elinb, tB, 0, 0, 0, hbuf, ME, nullptr, pWbuf, stream);

  float* hcur = hbuf;
  float* tC   = tCb;

  auto outblock = [&](const OutP& o, const float* h) {
    run_gemm<NRw, HC, false, false, E_CMUL>(rbf, o.rbfW, 0, 0, 0, 0, h, tA, ME, nullptr, pWbuf, stream);
    zero(nodeHC, MN * HC);
    long tot = (long)BSZ * NEDGE * (HC / 4);
    scatter_node_kernel<<<dim3((unsigned)((tot + 255) / 256)), dim3(256), 0, stream>>>(tA, iIdx, nodeHC);
    run_gemm<HC, OEw, false, false, 0>(nodeHC, o.upW, 0, 0, 0, 0, 0, nOE1, MN, nullptr, pWbuf, stream);
    run_gemm<OEw, OEw, false, false, E_BIAS | E_ACT>(nOE1, o.l0W, o.l0b, 0, 0, 0, 0, nOE2, MN, nullptr, pWbuf, stream);
    run_gemm<OEw, OEw, false, false, E_BIAS | E_ACT>(nOE2, o.l1W, o.l1b, 0, 0, 0, 0, nOE1, MN, nullptr, pWbuf, stream);
    run_gemm<OEw, OEw, false, false, E_BIAS | E_ACT>(nOE1, o.l2W, o.l2b, 0, 0, 0, 0, nOE2, MN, nullptr, pWbuf, stream);
    out_dot_kernel<<<dim3((unsigned)((MN + 255) / 256)), dim3(256), 0, stream>>>(nOE2, o.linW, (float*)d_out, (int)MN, OEw);
  };

  outblock(outp[0], hcur);

  for (int bi = 0; bi < NBLK; ++bi) {
    const BlkP& q = blk[bi];
    run_gemm<HC, HC, false, false, E_BIAS | E_ACT>(hcur, q.jiW, q.jib, 0, 0, 0, 0, tA, ME, nullptr, pWbuf, stream);
    run_gemm<HC, HC, false, false, E_BIAS | E_ACT>(hcur, q.kjW, q.kjb, 0, 0, 0, 0, tB, ME, nullptr, pWbuf, stream);
    run_gemm<NRw, BEw, false, false, 0>(rbf, q.rbf1W, 0, 0, 0, 0, 0, eBE, ME, nullptr, pWbuf, stream);
    run_gemm<BEw, HC, false, false, 0>(eBE, q.rbf2W, 0, 0, 0, 0, 0, tC, ME, nullptr, pWbuf, stream);
    run_gemm<HC, IEw, false, true, E_ACT>(tB, q.downW, 0, 0, 0, tC, 0, eIE1, ME, nullptr, pWbuf, stream);
    run_gemm<NSw * NRw, BEw, false, false, 0>(sbf, q.sbf1W, 0, 0, 0, 0, 0, tBE, MT, nullptr, pWbuf, stream);
    run_gemm<BEw, IEw, false, false, 0>(tBE, q.sbf2W, 0, 0, 0, 0, 0, tIE, MT, nullptr, pWbuf, stream);
    zero(eIE2, ME * IEw);
    long ttot = (long)BSZ * NTRIP * (IEw / 4);
    triplet_scatter_kernel<<<dim3((unsigned)((ttot + 255) / 256)), dim3(256), 0, stream>>>(eIE1, tIE, idx_kj, idx_ji, eIE2);
    run_gemm<IEw, HC, false, false, E_ACT | E_RES>(eIE2, q.upW, 0, 0, tA, 0, 0, tB, ME, nullptr, pWbuf, stream);
    run_gemm<HC, HC, false, false, E_BIAS | E_ACT>(tB, q.bl1W, q.bl1b, 0, 0, 0, 0, tC, ME, nullptr, pWbuf, stream);
    run_gemm<HC, HC, false, false, E_BIAS | E_ACT | E_RES>(tC, q.bl2W, q.bl2b, 0, tB, 0, 0, tB, ME, nullptr, pWbuf, stream);
    run_gemm<HC, HC, false, false, E_BIAS | E_ACT | E_RES>(tB, q.linW, q.linb, 0, hcur, 0, 0, tC, ME, nullptr, pWbuf, stream);
    run_gemm<HC, HC, false, false, E_BIAS | E_ACT>(tC, q.al1W, q.al1b, 0, 0, 0, 0, tA, ME, nullptr, pWbuf, stream);
    run_gemm<HC, HC, false, false, E_BIAS | E_ACT | E_RES>(tA, q.al2W, q.al2b, 0, tC, 0, 0, tC, ME, nullptr, pWbuf, stream);
    float* tmp = hcur; hcur = tC; tC = tmp;

    outblock(outp[bi + 1], hcur);
  }
}
